// GINE_84799834292736
// MI455X (gfx1250) — matrix operand order repaired and checked by reading, not compiled, not run on hardware
//
#include <hip/hip_runtime.h>
#include <hip/hip_bf16.h>

// ---------------------------------------------------------------------------
// GINE forward for MI455X (gfx1250, wave32, WMMA).
//   HIDDEN=128, EDGE_FEATS=16, 4 layers.
//   - ea = edge_attr @ We + be      (once, stored f16 in ws)
//   - per layer: zero agg; message gather+relu+atomic scatter;
//                fused WMMA MLP (GEMM-relu-GEMM) + LayerNorm + relu + residual
// ---------------------------------------------------------------------------

#define HID 128
#define MB  64   // node rows per MLP block

typedef __attribute__((ext_vector_type(16))) _Float16 v16h;
typedef __attribute__((ext_vector_type(2)))  _Float16 v2h;
typedef __attribute__((ext_vector_type(4)))  _Float16 v4h;
typedef __attribute__((ext_vector_type(8)))  float    v8f;

// ---- A fragment (16x32 f16, MxK): contiguous K order, k = 16*half + s ----------------
// lane r=lane&15 holds row M=r; half=lane>>4 selects K sub-blocks.
// VGPR pair i -> K pair p: half*8+i
__device__ __forceinline__ v16h load_a_frag(const _Float16* rowp, int half) {
  v16h a;
#pragma unroll
  for (int i = 0; i < 8; ++i) {
    int p = half * 8 + i;
    v2h t2 = *(const v2h*)(rowp + 2 * p);
    a[2 * i]     = t2[0];
    a[2 * i + 1] = t2[1];
  }
  return a;
}

// ---- B fragment (32x16 f16, KxN) from N-major (transposed) weights -------
// lane n=lane&15 is column N; half=lane>>4; VGPR i holds K=2i(+16*half).
// With Wt[n][k] layout this is 16 contiguous f16 = one 32B load.
__device__ __forceinline__ v16h load_b_frag(const _Float16* colp) {
  return *(const v16h*)colp;
}

// ===========================================================================
// Kernel: transpose + convert fp32 weights [L][K][N] -> f16 [L][N][K]
// ===========================================================================
__global__ __launch_bounds__(256) void wcvt_kernel(const float* __restrict__ W,
                                                   _Float16* __restrict__ Wt,
                                                   int total) {
  int idx = blockIdx.x * 256 + threadIdx.x;
  if (idx >= total) return;
  int l   = idx / (HID * HID);
  int rem = idx - l * HID * HID;
  int n   = rem >> 7;
  int k   = rem & 127;
  Wt[idx] = (_Float16)W[l * HID * HID + k * HID + n];
}

// ===========================================================================
// Kernel: ea[e][j] = edge_attr[e][:] @ We[:,j] + be[j], stored f16
// ===========================================================================
__global__ __launch_bounds__(256) void edge_embed_kernel(
    const float* __restrict__ edge_attr, const float* __restrict__ We,
    const float* __restrict__ be, _Float16* __restrict__ ea, int E) {
  __shared__ float sW[16 * HID];
  __shared__ float sB[HID];
  for (int i = threadIdx.x; i < 16 * HID; i += 256) sW[i] = We[i];
  if (threadIdx.x < HID) sB[threadIdx.x] = be[threadIdx.x];
  __syncthreads();

  const int j     = threadIdx.x & 127;
  const int esub  = threadIdx.x >> 7;  // 0..1
  const int ebase = blockIdx.x * 16;
  int eend = ebase + 16; if (eend > E) eend = E;
  for (int e = ebase + esub; e < eend; e += 2) {
    float acc = sB[j];
#pragma unroll
    for (int k = 0; k < 16; ++k) acc += edge_attr[e * 16 + k] * sW[k * HID + j];
    ea[(size_t)e * HID + j] = (_Float16)acc;
  }
}

// ===========================================================================
// Kernel: per-edge message m = relu(h[src] + ea); atomic scatter-add to agg
// One wave per edge; lane owns a contiguous float4 of the 128 features.
// ===========================================================================
__global__ __launch_bounds__(256) void message_kernel(
    const float* __restrict__ h, const _Float16* __restrict__ ea,
    const int* __restrict__ src, const int* __restrict__ dst,
    float* __restrict__ agg, int E) {
  int w = (blockIdx.x * blockDim.x + threadIdx.x) >> 5;
  if (w >= E) return;
  int lane = threadIdx.x & 31;
  int s = src[w];
  int d = dst[w];
  const float4 hv = *(const float4*)&h[(size_t)s * HID + lane * 4];
  const v4h    e4 = *(const v4h*)&ea[(size_t)w * HID + lane * 4];
  float m0 = fmaxf(hv.x + (float)e4[0], 0.f);
  float m1 = fmaxf(hv.y + (float)e4[1], 0.f);
  float m2 = fmaxf(hv.z + (float)e4[2], 0.f);
  float m3 = fmaxf(hv.w + (float)e4[3], 0.f);
  float* ap = &agg[(size_t)d * HID + lane * 4];
  atomicAdd(ap + 0, m0);
  atomicAdd(ap + 1, m1);
  atomicAdd(ap + 2, m2);
  atomicAdd(ap + 3, m3);
}

// ===========================================================================
// Kernel: fused MLP + LayerNorm + residual, WMMA f32_16x16x32_f16.
//   z0 = h + agg ; y = relu(z0@W1 + b1) ; z = y@W2 + b2
//   z = LN(z)*g + b ; h = relu(z) + h
// 256 threads = 8 waves; MB=64 rows; wave w -> m-tile w>>1, n-tiles (w&1)*4..+3
// ===========================================================================
__global__ __launch_bounds__(256) void mlp_fused_kernel(
    float* __restrict__ h, const float* __restrict__ agg,
    const _Float16* __restrict__ W1t, const _Float16* __restrict__ W2t,
    const float* __restrict__ b1, const float* __restrict__ b2,
    const float* __restrict__ lng, const float* __restrict__ lnb, int n) {
  __shared__ _Float16 sA[MB * HID];  // 16 KB: z0 tile, then y tile
  __shared__ float    sZ[MB * HID];  // 32 KB: z for LayerNorm
  __shared__ float    sMu[MB], sRs[MB];

  const int tid  = threadIdx.x;
  const int lane = tid & 31;
  const int wave = tid >> 5;
  const int row0 = blockIdx.x * MB;
  const int r16  = lane & 15;
  const int half = lane >> 4;
  const int mt   = wave >> 1;         // 0..3
  const int nt0  = (wave & 1) * 4;    // 0 or 4

  // ---- stage z0 = h + agg into LDS as f16 --------------------------------
  for (int i = tid; i < MB * HID; i += 256) {
    int r = i >> 7, c = i & 127;
    int gr = row0 + r;
    float v = 0.f;
    if (gr < n) v = h[(size_t)gr * HID + c] + agg[(size_t)gr * HID + c];
    sA[i] = (_Float16)v;
  }
  __syncthreads();

  // ---- GEMM1: y = relu(z0 @ W1 + b1) -------------------------------------
  v16h a[4];
#pragma unroll
  for (int ks = 0; ks < 4; ++ks)
    a[ks] = load_a_frag(&sA[(mt * 16 + r16) * HID + ks * 32], half);

  v8f acc[4];
#pragma unroll
  for (int j = 0; j < 4; ++j) {
    const int nt = nt0 + j;
    v8f c = {};
#pragma unroll
    for (int ks = 0; ks < 4; ++ks) {
      v16h b = load_b_frag(&W1t[(nt * 16 + r16) * HID + ks * 32 + half * 16]);
      c = __builtin_amdgcn_wmma_f32_16x16x32_f16(false, a[ks], false, b,
                                                 (short)0, c, false, false);
    }
    acc[j] = c;
  }
  __syncthreads();  // all waves finished reading sA as z0

  // ---- bias + relu -> y back to LDS (f16) --------------------------------
#pragma unroll
  for (int j = 0; j < 4; ++j) {
    const int col = (nt0 + j) * 16 + r16;   // C/D: lane&15 = N column
    const float bb = b1[col];
#pragma unroll
    for (int v = 0; v < 8; ++v) {
      int row = mt * 16 + half * 8 + v;     // C/D: VGPR v -> M = v + 8*half
      sA[row * HID + col] = (_Float16)fmaxf(acc[j][v] + bb, 0.f);
    }
  }
  __syncthreads();

  // ---- GEMM2: z = y @ W2 + b2 -> sZ (f32) --------------------------------
#pragma unroll
  for (int ks = 0; ks < 4; ++ks)
    a[ks] = load_a_frag(&sA[(mt * 16 + r16) * HID + ks * 32], half);

#pragma unroll
  for (int j = 0; j < 4; ++j) {
    const int nt = nt0 + j;
    v8f c = {};
#pragma unroll
    for (int ks = 0; ks < 4; ++ks) {
      v16h b = load_b_frag(&W2t[(nt * 16 + r16) * HID + ks * 32 + half * 16]);
      c = __builtin_amdgcn_wmma_f32_16x16x32_f16(false, a[ks], false, b,
                                                 (short)0, c, false, false);
    }
    const int col = nt * 16 + r16;
    const float bb = b2[col];
#pragma unroll
    for (int v = 0; v < 8; ++v) {
      int row = mt * 16 + half * 8 + v;
      sZ[row * HID + col] = c[v] + bb;
    }
  }
  __syncthreads();

  // ---- LayerNorm + relu + residual ---------------------------------------
  const int r  = tid >> 2;   // 0..63
  const int c0 = tid & 3;
  float s = 0.f, s2 = 0.f;
  for (int c = c0; c < HID; c += 4) {
    float z = sZ[r * HID + c];
    s += z; s2 += z * z;
  }
  s  += __shfl_xor(s, 1);  s2 += __shfl_xor(s2, 1);
  s  += __shfl_xor(s, 2);  s2 += __shfl_xor(s2, 2);
  if (c0 == 0) {
    float mu  = s * (1.f / HID);
    float var = s2 * (1.f / HID) - mu * mu;
    sMu[r] = mu;
    sRs[r] = rsqrtf(var + 1e-5f);
  }
  __syncthreads();

  const int gr = row0 + r;
  if (gr < n) {
    const float mu = sMu[r], rs = sRs[r];
    for (int c = c0; c < HID; c += 4) {
      float z = (sZ[r * HID + c] - mu) * rs * lng[c] + lnb[c];
      size_t gi = (size_t)gr * HID + c;
      h[gi] = fmaxf(z, 0.f) + h[gi];
    }
  }
}

// ===========================================================================
// small utility kernels
// ===========================================================================
__global__ __launch_bounds__(256) void copy_kernel(const float* __restrict__ in,
                                                   float* __restrict__ out, int n) {
  int i = blockIdx.x * 256 + threadIdx.x;
  if (i < n) out[i] = in[i];
}

__global__ __launch_bounds__(256) void zero_kernel(float* __restrict__ p, int n) {
  int i = blockIdx.x * 256 + threadIdx.x;
  if (i < n) p[i] = 0.f;
}

// ===========================================================================
// launch
// ===========================================================================
extern "C" void kernel_launch(void* const* d_in, const int* in_sizes, int n_in,
                              void* d_out, int out_size, void* d_ws, size_t ws_size,
                              hipStream_t stream) {
  const float* x         = (const float*)d_in[0];
  // d_in[1] = batch_index (unused by reference output)
  const int*   ei        = (const int*)d_in[2];     // [2,E]: row0=src, row1=dst
  const float* edge_attr = (const float*)d_in[3];
  const float* We        = (const float*)d_in[4];
  const float* be        = (const float*)d_in[5];
  const float* W1        = (const float*)d_in[6];
  const float* b1        = (const float*)d_in[7];
  const float* W2        = (const float*)d_in[8];
  const float* b2        = (const float*)d_in[9];
  const float* lng       = (const float*)d_in[10];
  const float* lnb       = (const float*)d_in[11];

  const int N = in_sizes[0] / HID;           // 50000
  const int E = in_sizes[2] / 2;             // 640000
  const int L = in_sizes[6] / (HID * HID);   // 4

  // workspace carve-up
  char* ws = (char*)d_ws;
  _Float16* ea = (_Float16*)ws;  ws += (size_t)E * HID * sizeof(_Float16);
  float*   agg = (float*)ws;     ws += (size_t)N * HID * sizeof(float);
  _Float16* W1t = (_Float16*)ws; ws += (size_t)L * HID * HID * sizeof(_Float16);
  _Float16* W2t = (_Float16*)ws;

  float* h = (float*)d_out;  // h lives in the output buffer across layers

  const int wtot = L * HID * HID;
  wcvt_kernel<<<(wtot + 255) / 256, 256, 0, stream>>>(W1, W1t, wtot);
  wcvt_kernel<<<(wtot + 255) / 256, 256, 0, stream>>>(W2, W2t, wtot);

  edge_embed_kernel<<<(E + 15) / 16, 256, 0, stream>>>(edge_attr, We, be, ea, E);

  copy_kernel<<<((size_t)N * HID + 255) / 256, 256, 0, stream>>>(x, h, N * HID);

  for (int l = 0; l < L; ++l) {
    zero_kernel<<<((size_t)N * HID + 255) / 256, 256, 0, stream>>>(agg, N * HID);
    message_kernel<<<((size_t)E * 32 + 255) / 256, 256, 0, stream>>>(
        h, ea, ei, ei + E, agg, E);
    mlp_fused_kernel<<<(N + MB - 1) / MB, 256, 0, stream>>>(
        h, agg, W1t + (size_t)l * HID * HID, W2t + (size_t)l * HID * HID,
        b1 + l * HID, b2 + l * HID, lng + l * HID, lnb + l * HID, N);
  }
}
